// AHFAN_88854283419927
// MI455X (gfx1250) — matrix operand order repaired and checked by reading, not compiled, not run on hardware
//
#include <hip/hip_runtime.h>
#include <hip/hip_bf16.h>
#include <math.h>

#define EPSV 1e-12f

typedef __attribute__((ext_vector_type(16))) _Float16 v16h;
typedef __attribute__((ext_vector_type(8)))  _Float16 v8h;
typedef __attribute__((ext_vector_type(8)))  float    v8f;

__device__ __forceinline__ float wave_sum(float v) {
    // wave32 butterfly reduction; result valid in all lanes
    #pragma unroll
    for (int m = 16; m >= 1; m >>= 1) v += __shfl_xor(v, m, 32);
    return v;
}

__device__ __forceinline__ float fast_tanh(float x) {
    // 1 - 2/(e^{2x}+1); saturates correctly for |x| large (exp->inf or 0)
    return 1.0f - 2.0f / (__expf(2.0f * x) + 1.0f);
}

// ---------------------------------------------------------------------------
// Shared helpers for the WMMA GEMM family.
// Block = 32 rows, 8 waves: wave w -> row-half (w>>2), col tile (w&3).
// W pre-packed in LDS in WMMA B-fragment order: fragment read = 32B contiguous.
// rows MUST be a multiple of 32 (true here: N=100000, 2N=200000).
// ---------------------------------------------------------------------------
__device__ __forceinline__ void stage_A(const float* __restrict__ in, int row0,
                                        _Float16* __restrict__ Ah, int tid) {
    const int i0 = tid * 8;                 // 256*8 = 2048 = 32*64 halfs
    const int r  = i0 >> 6, c = i0 & 63;
    const float* p = in + (size_t)(row0 + r) * 64 + c;
    v8h av;
    #pragma unroll
    for (int j = 0; j < 8; ++j) av[j] = (_Float16)p[j];
    *(v8h*)&Ah[i0] = av;
}

// pack one B-fragment 8-half granule; task = {kc,jh,wv,lane}
__device__ __forceinline__ void stage_B_task(const float* __restrict__ W,
                                             _Float16* __restrict__ Bp, int task) {
    const int lane = task & 31;
    const int wv   = (task >> 5) & 3;
    const int jh   = (task >> 7) & 1;
    const int kc   = (task >> 8) & 1;
    const int col    = wv * 16 + (lane & 15);
    const int kstart = kc * 32 + (lane >> 4) * 8 + jh * 16;
    v8h bv;
    #pragma unroll
    for (int i = 0; i < 8; ++i) bv[i] = (_Float16)W[(kstart + i) * 64 + col];
    *(v8h*)&Bp[((kc * 4 + wv) * 32 + lane) * 16 + jh * 8] = bv;
}

__device__ __forceinline__ v8f wmma_tile(const _Float16* __restrict__ Ah,
                                         const _Float16* __restrict__ Bp,
                                         int arow, int wv, int hi) {
    v8f acc = {};
    #pragma unroll
    for (int kc = 0; kc < 2; ++kc) {
        // A 16x32 f16 layout: lanes 0-15 hold K 0..7/16..23, lanes 16-31 K 8..15/24..31
        const int ka = kc * 32 + hi * 8;
        v16h a, b;
        *((v8h*)&a)     = *(const v8h*)&Ah[arow * 64 + ka];
        *((v8h*)&a + 1) = *(const v8h*)&Ah[arow * 64 + ka + 16];
        const int lane = (threadIdx.x & 31);
        b = *(const v16h*)&Bp[((kc * 4 + wv) * 32 + lane) * 16];
        acc = __builtin_amdgcn_wmma_f32_16x16x32_f16(
            false, a, false, b, (short)0, acc, false, false);
    }
    return acc;
}

// ---------------------------------------------------------------------------
// Generic GEMM: out[rows,64] = act(in @ W (+ bias)); ACT: 0=none 1=relu 2=tanh
// ---------------------------------------------------------------------------
template<int ACT, bool HAS_BIAS>
__global__ __launch_bounds__(256)
void gemm64_kernel(const float* __restrict__ in, const float* __restrict__ W,
                   const float* __restrict__ bias, float* __restrict__ out,
                   int rows) {
    __shared__ __align__(16) _Float16 Ah[32 * 64];            // 4 KB
    __shared__ __align__(16) _Float16 Bp[2 * 4 * 32 * 16];    // 8 KB
    const int tid  = threadIdx.x;
    const int row0 = blockIdx.x * 32;
    (void)rows;

    stage_A(in, row0, Ah, tid);
    #pragma unroll
    for (int t = 0; t < 2; ++t) stage_B_task(W, Bp, tid + t * 256);
    __syncthreads();

    const int lane = tid & 31;
    const int w    = tid >> 5;
    const int wv   = w & 3;
    const int rh   = w >> 2;
    const int mn   = lane & 15;
    const int hi   = lane >> 4;

    v8f acc = wmma_tile(Ah, Bp, rh * 16 + mn, wv, hi);

    const int col   = wv * 16 + mn;
    const float bv  = HAS_BIAS ? bias[col] : 0.0f;
    const int rbase = row0 + rh * 16 + hi * 8;   // C/D: lanes<16 -> M=r, else M=8+r
    #pragma unroll
    for (int r = 0; r < 8; ++r) {
        float v = acc[r] + bv;
        if (ACT == 1)      v = fmaxf(v, 0.0f);
        else if (ACT == 2) v = fast_tanh(v);
        out[(size_t)(rbase + r) * 64 + col] = v;
    }
}

// ---------------------------------------------------------------------------
// Fused GCN transform: computes hw1 = h@wg1, hw2 = h@wg2 AND the self-loop
// initialized filters hf[n][k][:] = dinv[n]^2 * hw_k[n][:] + bg_k.
// A staged once, both W matrices packed in LDS, 4 WMMAs per wave.
// ---------------------------------------------------------------------------
__global__ __launch_bounds__(256)
void gemm64_gcn_kernel(const float* __restrict__ in,
                       const float* __restrict__ W1, const float* __restrict__ W2,
                       const float* __restrict__ bg1, const float* __restrict__ bg2,
                       const float* __restrict__ dinv,
                       float* __restrict__ hw1, float* __restrict__ hw2,
                       float* __restrict__ hf, int rows) {
    __shared__ __align__(16) _Float16 Ah[32 * 64];                // 4 KB
    __shared__ __align__(16) _Float16 Bp[2][2 * 4 * 32 * 16];     // 16 KB
    const int tid  = threadIdx.x;
    const int row0 = blockIdx.x * 32;
    (void)rows;

    stage_A(in, row0, Ah, tid);
    #pragma unroll
    for (int t = 0; t < 4; ++t) {
        const int task = tid + t * 256;          // 1024 tasks; bit9 = matrix
        const int mat  = (task >> 9) & 1;
        stage_B_task(mat ? W2 : W1, Bp[mat], task & 511);
    }
    __syncthreads();

    const int lane = tid & 31;
    const int w    = tid >> 5;
    const int wv   = w & 3;
    const int rh   = w >> 2;
    const int mn   = lane & 15;
    const int hi   = lane >> 4;
    const int arow = rh * 16 + mn;

    v8f acc1 = wmma_tile(Ah, Bp[0], arow, wv, hi);
    v8f acc2 = wmma_tile(Ah, Bp[1], arow, wv, hi);

    const int col   = wv * 16 + mn;
    const float bv1 = bg1[col];
    const float bv2 = bg2[col];
    const int rbase = row0 + rh * 16 + hi * 8;
    #pragma unroll
    for (int r = 0; r < 8; ++r) {
        const int row = rbase + r;
        float dv = dinv[row];
        float sn = dv * dv;
        float v1 = acc1[r], v2 = acc2[r];
        hw1[(size_t)row * 64 + col] = v1;
        hw2[(size_t)row * 64 + col] = v2;
        hf[(size_t)row * 128 + col]      = sn * v1 + bv1;   // self-loop + bias
        hf[(size_t)row * 128 + 64 + col] = sn * v2 + bv2;
    }
}

// ---------------------------------------------------------------------------
// Degree / normalization
// ---------------------------------------------------------------------------
__global__ void init_deg_kernel(float* deg, int n_nodes) {
    int i = blockIdx.x * blockDim.x + threadIdx.x;
    if (i < n_nodes) deg[i] = 1.0f;          // self-loop contribution
}
__global__ void deg_scatter_kernel(const int* __restrict__ ei, float* deg, int nE) {
    int e = blockIdx.x * blockDim.x + threadIdx.x;
    if (e < nE) atomicAdd(&deg[ei[nE + e]], 1.0f);   // col = ei[1][e]
}
__global__ void dinv_kernel(const float* __restrict__ deg, float* dinv, int n_nodes) {
    int i = blockIdx.x * blockDim.x + threadIdx.x;
    if (i < n_nodes) {
        float d = deg[i];
        dinv[i] = (d > 0.0f) ? rsqrtf(d) : 0.0f;
    }
}

__global__ __launch_bounds__(256)
void gcn_scatter_kernel(const int* __restrict__ ei,
                        const float* __restrict__ dinv,
                        const float* __restrict__ hw1,
                        const float* __restrict__ hw2,
                        float* __restrict__ hf, int nE) {
    int e    = blockIdx.x * 8 + (threadIdx.x >> 5);
    int lane = threadIdx.x & 31;
    if (e >= nE) return;
    int r = ei[e], c = ei[nE + e];
    float nm = dinv[r] * dinv[c];
    const int h = lane * 2;
    float2 m1 = *(const float2*)&hw1[(size_t)r * 64 + h];
    float2 m2 = *(const float2*)&hw2[(size_t)r * 64 + h];
    atomicAdd(&hf[(size_t)c * 128 + h],          nm * m1.x);
    atomicAdd(&hf[(size_t)c * 128 + h + 1],      nm * m1.y);
    atomicAdd(&hf[(size_t)c * 128 + 64 + h],     nm * m2.x);
    atomicAdd(&hf[(size_t)c * 128 + 64 + h + 1], nm * m2.y);
}

// ---------------------------------------------------------------------------
// AGNN: row-normalize, per-edge cosine attention, segment softmax (no max
// subtraction needed: |alpha| <= |beta|), weighted aggregation.
// ---------------------------------------------------------------------------
__global__ __launch_bounds__(256)
void xn_kernel(const float* __restrict__ h, const float* __restrict__ beta_p,
               float* __restrict__ xn, float* __restrict__ esl,
               float* __restrict__ s, int n_nodes) {
    int n    = blockIdx.x * 8 + (threadIdx.x >> 5);
    int lane = threadIdx.x & 31;
    if (n >= n_nodes) return;
    float a0 = h[(size_t)n * 64 + lane];
    float a1 = h[(size_t)n * 64 + 32 + lane];
    float ss = wave_sum(a0 * a0 + a1 * a1);
    float inv = 1.0f / (sqrtf(ss) + EPSV);
    xn[(size_t)n * 64 + lane]      = a0 * inv;
    xn[(size_t)n * 64 + 32 + lane] = a1 * inv;
    if (lane == 0) {
        float selfdot = ss * inv * inv;          // dot(xn, xn)
        float es = __expf(beta_p[0] * selfdot);
        esl[n] = es;
        s[n]   = es;                              // softmax denom init (self edge)
    }
}

__global__ __launch_bounds__(256)
void agnn_alpha_kernel(const int* __restrict__ ei, const float* __restrict__ xn,
                       const float* __restrict__ beta_p,
                       float* __restrict__ expa, float* __restrict__ s, int nE) {
    int e    = blockIdx.x * 8 + (threadIdx.x >> 5);
    int lane = threadIdx.x & 31;
    if (e >= nE) return;
    int r = ei[e], c = ei[nE + e];
    float d = xn[(size_t)r * 64 + lane]      * xn[(size_t)c * 64 + lane]
            + xn[(size_t)r * 64 + 32 + lane] * xn[(size_t)c * 64 + 32 + lane];
    d = wave_sum(d);
    float ea = __expf(beta_p[0] * d);
    if (lane == 0) {
        expa[e] = ea;
        atomicAdd(&s[c], ea);
    }
}

__global__ void h1_init_kernel(const float* __restrict__ h,
                               const float* __restrict__ esl,
                               const float* __restrict__ s,
                               float* __restrict__ h1, int n_nodes) {
    int i = blockIdx.x * blockDim.x + threadIdx.x;
    if (i < n_nodes * 64) {
        int n = i >> 6;
        h1[i] = (esl[n] / s[n]) * h[i];
    }
}

__global__ __launch_bounds__(256)
void agnn_aggr_kernel(const int* __restrict__ ei, const float* __restrict__ expa,
                      const float* __restrict__ s, const float* __restrict__ h,
                      float* __restrict__ h1, int nE) {
    int e    = blockIdx.x * 8 + (threadIdx.x >> 5);
    int lane = threadIdx.x & 31;
    if (e >= nE) return;
    int r = ei[e], c = ei[nE + e];
    float w = expa[e] / s[c];
    const int hh = lane * 2;
    float2 m = *(const float2*)&h[(size_t)r * 64 + hh];
    atomicAdd(&h1[(size_t)c * 64 + hh],     w * m.x);
    atomicAdd(&h1[(size_t)c * 64 + hh + 1], w * m.y);
}

// ---------------------------------------------------------------------------
// Fusion head: logits = <tanh(hf@wf+bf), tanh(h@wx+bx)>, softmax over k=0,1,
// res = sum_k score_k * hf_k, y = [res | h1] @ wc + bc.  One wave per node.
// ---------------------------------------------------------------------------
__global__ __launch_bounds__(256)
void final_kernel(const float* __restrict__ hf, const float* __restrict__ hproj,
                  const float* __restrict__ xproj, const float* __restrict__ h1,
                  const float* __restrict__ wc, const float* __restrict__ bc,
                  float* __restrict__ y, int n_nodes) {
    int n    = blockIdx.x * 8 + (threadIdx.x >> 5);
    int lane = threadIdx.x & 31;
    if (n >= n_nodes) return;
    float xp0 = xproj[(size_t)n * 64 + lane];
    float xp1 = xproj[(size_t)n * 64 + 32 + lane];
    size_t p0 = (size_t)(2 * n) * 64, p1 = (size_t)(2 * n + 1) * 64;
    float l0 = wave_sum(hproj[p0 + lane] * xp0 + hproj[p0 + 32 + lane] * xp1);
    float l1 = wave_sum(hproj[p1 + lane] * xp0 + hproj[p1 + 32 + lane] * xp1);
    float mx = fmaxf(l0, l1);
    float e0 = __expf(l0 - mx), e1 = __expf(l1 - mx);
    float inv = 1.0f / (e0 + e1);
    float s0 = e0 * inv, s1 = e1 * inv;
    size_t hb = (size_t)n * 128;
    float r0 = hf[hb + lane]      * s0 + hf[hb + 64 + lane] * s1;  // h = lane
    float r1 = hf[hb + 32 + lane] * s0 + hf[hb + 96 + lane] * s1;  // h = lane+32
    float h10 = h1[(size_t)n * 64 + lane];
    float h11 = h1[(size_t)n * 64 + 32 + lane];
    float a0 = r0 * wc[lane * 2]         + r1 * wc[(lane + 32) * 2]
             + h10 * wc[(64 + lane) * 2] + h11 * wc[(96 + lane) * 2];
    float a1 = r0 * wc[lane * 2 + 1]         + r1 * wc[(lane + 32) * 2 + 1]
             + h10 * wc[(64 + lane) * 2 + 1] + h11 * wc[(96 + lane) * 2 + 1];
    a0 = wave_sum(a0);
    a1 = wave_sum(a1);
    if (lane == 0) {
        y[(size_t)n * 2]     = a0 + bc[0];
        y[(size_t)n * 2 + 1] = a1 + bc[1];
    }
}

// ---------------------------------------------------------------------------
extern "C" void kernel_launch(void* const* d_in, const int* in_sizes, int n_in,
                              void* d_out, int out_size, void* d_ws, size_t ws_size,
                              hipStream_t stream) {
    (void)n_in; (void)out_size; (void)ws_size;
    const float* x    = (const float*)d_in[0];
    const int*   ei   = (const int*)d_in[1];
    const float* w1   = (const float*)d_in[2];
    const float* b1   = (const float*)d_in[3];
    const float* w2   = (const float*)d_in[4];
    const float* b2   = (const float*)d_in[5];
    const float* w3   = (const float*)d_in[6];
    const float* b3   = (const float*)d_in[7];
    const float* wg1  = (const float*)d_in[8];
    const float* bg1  = (const float*)d_in[9];
    const float* wg2  = (const float*)d_in[10];
    const float* bg2  = (const float*)d_in[11];
    const float* beta = (const float*)d_in[12];
    const float* wf   = (const float*)d_in[13];
    const float* bf   = (const float*)d_in[14];
    const float* wx   = (const float*)d_in[15];
    const float* bx   = (const float*)d_in[16];
    const float* wc   = (const float*)d_in[17];
    const float* bc   = (const float*)d_in[18];
    float* y = (float*)d_out;

    const int N = in_sizes[0] / 64;    // 100000 (multiple of 32)
    const int E = in_sizes[1] / 2;
    const size_t NH = (size_t)N * 64;

    float* ws = (float*)d_ws;
    size_t off = 0;
    float* H     = ws + off; off += NH;        // h (MLP output)
    float* T     = ws + off; off += NH;        // MLP ping buffer
    float* HW1   = ws + off; off += NH;        // h@wg1 ; reused as xproj
    float* HW2   = ws + off; off += NH;        // h@wg2 ; reused as h1
    float* XN    = ws + off; off += NH;        // normalized h
    float* HF    = ws + off; off += 2 * NH;    // [N,2,64] GCN filters
    float* HPROJ = ws + off; off += 2 * NH;    // tanh(hf@wf+bf)
    float* EXPA  = ws + off; off += (size_t)E; // per-edge exp(alpha)
    float* DEG   = ws + off; off += (size_t)N;
    float* DINV  = ws + off; off += (size_t)N;
    float* S     = ws + off; off += (size_t)N; // softmax denominators
    float* ESL   = ws + off; off += (size_t)N; // self-edge exp(alpha)
    float* XPROJ = HW1;                        // reuse after gcn_scatter
    float* H1    = HW2;                        // reuse after gcn_scatter

    const int nodeBlocks = (N + 255) / 256;
    const int nhBlocks   = (N * 64 + 255) / 256;
    const int edgeBlocks = (E + 255) / 256;
    const int nodeWave   = (N + 7) / 8;        // 1 wave per node, 8 waves/block
    const int edgeWave   = (E + 7) / 8;        // 1 wave per edge
    const int gR_N  = N / 32;                  // rows are multiples of 32
    const int gR_2N = (2 * N) / 32;

    // degree + symmetric normalization (self-loops folded in)
    init_deg_kernel<<<nodeBlocks, 256, 0, stream>>>(DEG, N);
    deg_scatter_kernel<<<edgeBlocks, 256, 0, stream>>>(ei, DEG, E);
    dinv_kernel<<<nodeBlocks, 256, 0, stream>>>(DEG, DINV, N);

    // input MLP (WMMA GEMMs)
    gemm64_kernel<1, true ><<<gR_N, 256, 0, stream>>>(x, w1, b1, H, N);
    gemm64_kernel<1, true ><<<gR_N, 256, 0, stream>>>(H, w2, b2, T, N);
    gemm64_kernel<0, true ><<<gR_N, 256, 0, stream>>>(T, w3, b3, H, N);

    // fused GCN transforms + self-loop/bias init of hf, then edge scatter
    gemm64_gcn_kernel<<<gR_N, 256, 0, stream>>>(H, wg1, wg2, bg1, bg2, DINV,
                                                HW1, HW2, HF, N);
    gcn_scatter_kernel<<<edgeWave, 256, 0, stream>>>(ei, DINV, HW1, HW2, HF, E);

    // AGNN attention
    xn_kernel<<<nodeWave, 256, 0, stream>>>(H, beta, XN, ESL, S, N);
    agnn_alpha_kernel<<<edgeWave, 256, 0, stream>>>(ei, XN, beta, EXPA, S, E);
    h1_init_kernel<<<nhBlocks, 256, 0, stream>>>(H, ESL, S, H1, N);
    agnn_aggr_kernel<<<edgeWave, 256, 0, stream>>>(ei, EXPA, S, H, H1, E);

    // fusion head projections (WMMA GEMMs) + attention-weighted classify
    gemm64_kernel<2, true ><<<gR_2N, 256, 0, stream>>>(HF, wf, bf, HPROJ, 2 * N);
    gemm64_kernel<2, true ><<<gR_N,  256, 0, stream>>>(H, wx, bx, XPROJ, N);
    final_kernel<<<nodeWave, 256, 0, stream>>>(HF, HPROJ, XPROJ, H1, wc, bc, y, N);
}
